// GM_GCN2_81028853006976
// MI455X (gfx1250) — matrix operand order repaired and checked by reading, not compiled, not run on hardware
//
#include <hip/hip_runtime.h>
#include <hip/hip_bf16.h>
#include <math.h>

typedef __bf16 v16bf __attribute__((ext_vector_type(16)));
typedef float  v8f   __attribute__((ext_vector_type(8)));

// Relaxed agent-scope f32 add: lowers to global_atomic_add_f32 (no return).
__device__ __forceinline__ void atomic_add_f32(float* p, float v) {
    __hip_atomic_fetch_add(p, v, __ATOMIC_RELAXED, __HIP_MEMORY_SCOPE_AGENT);
}

// ---------------------------------------------------------------------------
// Degree / normalization kernels (integer degree -> native u32 atomics)
// ---------------------------------------------------------------------------
__global__ void fill_ones_kernel(int* __restrict__ deg, int n) {
    int t = blockIdx.x * blockDim.x + threadIdx.x;
    if (t < n) deg[t] = 1;                       // self-loop contribution
}

__global__ void deg_edges_kernel(const int* __restrict__ col,
                                 int* __restrict__ deg, int e) {
    int t = blockIdx.x * blockDim.x + threadIdx.x;
    if (t < e) atomicAdd(&deg[col[t]], 1);       // global_atomic_add_u32
}

__global__ void deg_to_dinv_kernel(const int* __restrict__ deg,
                                   float* __restrict__ dinv, int n) {
    int t = blockIdx.x * blockDim.x + threadIdx.x;
    if (t < n) {
        float v = (float)deg[t];
        dinv[t] = (v > 0.0f) ? rsqrtf(v) : 0.0f;
    }
}

// ---------------------------------------------------------------------------
// SpMM: out = D^-1/2 (A + I) D^-1/2 @ h     (h, out are [N][64] f32)
// ---------------------------------------------------------------------------
// Self-loop term fully initializes `out` (no reliance on prior buffer state).
__global__ void spmm_self_kernel(const float* __restrict__ h,
                                 const float* __restrict__ dinv,
                                 float* __restrict__ out, int n16) {
    int t = blockIdx.x * blockDim.x + threadIdx.x;
    if (t < n16) {
        int v = t >> 4;                          // 16 float4 per node (64 feats)
        float s = dinv[v]; s = s * s;
        float4 hv = ((const float4*)h)[t];
        float4 o;
        o.x = s * hv.x; o.y = s * hv.y; o.z = s * hv.z; o.w = s * hv.w;
        ((float4*)out)[t] = o;
    }
}

// One wave (32 lanes) per edge; each lane handles 2 consecutive features.
__global__ void spmm_edges_kernel(const int* __restrict__ row,
                                  const int* __restrict__ col,
                                  const float* __restrict__ dinv,
                                  const float* __restrict__ h,
                                  float* __restrict__ out, int ne) {
    int gid  = blockIdx.x * blockDim.x + threadIdx.x;
    int e    = gid >> 5;
    int lane = gid & 31;
    if (e < ne) {
        int u = row[e], v = col[e];
        float w = dinv[u] * dinv[v];
        const float2 hv = *(const float2*)(h + (size_t)u * 64 + lane * 2);
        float* o = out + (size_t)v * 64 + lane * 2;
        atomic_add_f32(o,     w * hv.x);
        atomic_add_f32(o + 1, w * hv.y);
    }
}

// ---------------------------------------------------------------------------
// Elementwise relu copy (layer-0: x0 = h = relu(prop(h)))
// ---------------------------------------------------------------------------
__global__ void relu_copy_kernel(const float* __restrict__ in,
                                 float* __restrict__ out, int n4) {
    int t = blockIdx.x * blockDim.x + threadIdx.x;
    if (t < n4) {
        float4 v = ((const float4*)in)[t];
        v.x = fmaxf(v.x, 0.f); v.y = fmaxf(v.y, 0.f);
        v.z = fmaxf(v.z, 0.f); v.w = fmaxf(v.w, 0.f);
        ((float4*)out)[t] = v;
    }
}

// ---------------------------------------------------------------------------
// WMMA GEMM with (compile-time) fused GCNII combine:
//   Aeff = FUSE ? ca*A + cb*A2 : A
//   C    = act( [RESID: c_resid*Aeff] + c_acc*(Aeff@W) + [BIASED: bias] )
//   A,A2: [M][lda] f32, W: [K][wcols] f32 (staged to LDS as bf16, transposed)
//   K = KSTEPS*32, padded N = NT*16. One wave -> 16 x (NT*16) output tile.
// All feature flags are template parameters so every instantiation is
// straight-line code: A/A2 fragment loads coalesce into global_load_b128
// clauses, no per-element exec masking.
// ---------------------------------------------------------------------------
template<int NT, int KSTEPS, bool FUSE, bool RESID, bool BIASED, bool RELU>
__global__ __launch_bounds__(256) void gemm_wmma_kernel(
    const float* __restrict__ A, const float* __restrict__ A2,
    float ca, float cb, int lda, int M,
    const float* __restrict__ W, int wcols,
    const float* __restrict__ bias,
    float c_resid, float c_acc,
    float* __restrict__ C, int ldc, int ocols)
{
    constexpr int K    = KSTEPS * 32;
    constexpr int NPAD = NT * 16;
    constexpr int LROW = K + 16;          // +32B row pad: keeps 32B alignment,
                                          // skews LDS banks across columns
    __shared__ __align__(32) __bf16 sW[NPAD * LROW];

    // Cooperative stage of W into LDS: transposed [n][k], zero-padded cols.
    for (int t = threadIdx.x; t < NPAD * K; t += 256) {
        int n = t / K, k = t - n * K;
        float v = (n < wcols) ? W[(size_t)k * wcols + n] : 0.0f;
        sW[n * LROW + k] = (__bf16)v;
    }
    __syncthreads();

    int wave = threadIdx.x >> 5;
    int lane = threadIdx.x & 31;
    int mt   = blockIdx.x * 8 + wave;
    int mtiles = (M + 15) >> 4;
    if (mt >= mtiles) return;
    int mbase = mt << 4;

    // --- A fragment addressing (16-bit A 16x32 layout) ---
    int rowA = mbase + (lane & 15);
    if (rowA >= M) rowA = M - 1;          // clamp (M is a multiple of 16 here)
    const float* Ar  = A + (size_t)rowA * lda;
    const float* A2r = FUSE ? (A2 + (size_t)rowA * lda) : nullptr;
    const int kbA = (lane & 16) ? 16 : 0; // K-base within 32-chunk for A
    const int kbB = (lane & 16) ? 16 : 0; // K-base within 32-chunk for B

    v8f acc[NT];
#pragma unroll
    for (int i = 0; i < NT; ++i)
#pragma unroll
        for (int j = 0; j < 8; ++j) acc[i][j] = 0.0f;

#pragma unroll
    for (int ks = 0; ks < KSTEPS; ++ks) {
        const int k0 = ks * 32;
        v16bf a;
#pragma unroll
        for (int e2 = 0; e2 < 8; ++e2) {
            int i0 = k0 + kbA + e2;
            int i1 = k0 + kbA + 8 + e2;
            float x0v = Ar[i0], x1v = Ar[i1];
            if (FUSE) {
                x0v = ca * x0v + cb * A2r[i0];
                x1v = ca * x1v + cb * A2r[i1];
            }
            a[e2]     = (__bf16)x0v;
            a[8 + e2] = (__bf16)x1v;
        }
#pragma unroll
        for (int nt = 0; nt < NT; ++nt) {
            // B fragment: column n = nt*16 + (lane&15), 16 contiguous K values
            const __bf16* p = sW + (size_t)(nt * 16 + (lane & 15)) * LROW + k0 + kbB;
            v16bf b = *(const v16bf*)p;
            acc[nt] = __builtin_amdgcn_wmma_f32_16x16x32_bf16(
                false, a, false, b, (short)0, acc[nt], false, false);
        }
    }

    // --- epilogue (32-bit C/D 16x16 layout: VGPR r -> M = r (+8 hi lanes)) ---
    int mrow0 = mbase + ((lane & 16) ? 8 : 0);
    int ncl   = lane & 15;
#pragma unroll
    for (int nt = 0; nt < NT; ++nt) {
        int c = nt * 16 + ncl;
        if (c < ocols) {
#pragma unroll
            for (int r = 0; r < 8; ++r) {
                int mr = mrow0 + r;
                if (mr < M) {
                    float v = c_acc * acc[nt][r];
                    if (BIASED) v += bias[c];
                    if (RESID) {
                        float rv = A[(size_t)mr * lda + c];
                        if (FUSE) rv = ca * rv + cb * A2[(size_t)mr * lda + c];
                        v += c_resid * rv;
                    }
                    if (RELU) v = fmaxf(v, 0.0f);
                    C[(size_t)mr * ldc + c] = v;
                }
            }
        }
    }
}

// ---------------------------------------------------------------------------
extern "C" void kernel_launch(void* const* d_in, const int* in_sizes, int n_in,
                              void* d_out, int out_size, void* d_ws, size_t ws_size,
                              hipStream_t stream) {
    const float* x     = (const float*)d_in[0];
    const int*   eidx  = (const int*)d_in[1];
    const float* fc0_w = (const float*)d_in[2];
    const float* fc0_b = (const float*)d_in[3];
    const float* fc1_w = (const float*)d_in[4];
    const float* fc1_b = (const float*)d_in[5];
    const float* convw = (const float*)d_in[6];

    const int N = in_sizes[0] / 128;
    const int E = in_sizes[1] / 2;
    const int* row = eidx;
    const int* col = eidx + E;

    // scratch carve-out
    char* ws = (char*)d_ws;
    size_t off = 0;
    auto carve = [&](size_t bytes) {
        void* p = ws + off;
        off = (off + bytes + 255) & ~(size_t)255;
        return p;
    };
    int*   degi = (int*)carve((size_t)N * 4);
    float* dinv = (float*)carve((size_t)N * 4);
    float* hbuf = (float*)carve((size_t)N * 64 * 4);
    float* mbuf = (float*)carve((size_t)N * 64 * 4);
    float* x0   = (float*)carve((size_t)N * 64 * 4);
    float* out  = (float*)d_out;

    const int TB = 256;
    const int n16 = N * 16;                // N*64 floats as float4
    const int ethreads = E * 32;           // one wave per edge
    const int mtiles  = (N + 15) / 16;
    const int gblocks = (mtiles + 7) / 8;

    // 1) gcn_norm (integer degree, then rsqrt)
    fill_ones_kernel  <<<(N + TB - 1) / TB, TB, 0, stream>>>(degi, N);
    deg_edges_kernel  <<<(E + TB - 1) / TB, TB, 0, stream>>>(col, degi, E);
    deg_to_dinv_kernel<<<(N + TB - 1) / TB, TB, 0, stream>>>(degi, dinv, N);

    // 2) h = relu(x @ fc0_w + fc0_b)
    gemm_wmma_kernel<4, 4, false, false, true, true><<<gblocks, TB, 0, stream>>>(
        x, nullptr, 1.0f, 0.0f, 128, N, fc0_w, 64, fc0_b,
        0.0f, 1.0f, hbuf, 64, 64);

    // 3) x0 = h = relu(prop(h))
    spmm_self_kernel <<<(n16 + TB - 1) / TB, TB, 0, stream>>>(hbuf, dinv, mbuf, n16);
    spmm_edges_kernel<<<(ethreads + TB - 1) / TB, TB, 0, stream>>>(row, col, dinv, hbuf, mbuf, E);
    relu_copy_kernel <<<(n16 + TB - 1) / TB, TB, 0, stream>>>(mbuf, x0, n16);

    // 4) GCNII layers i = 1..3 with fused combine:
    //    m = 0.9*prop(h) + 0.1*x0 ;  h = relu((1-beta)*m + beta*(m@W_i))
    const float* h = x0;
    for (int i = 1; i < 4; ++i) {
        spmm_self_kernel <<<(n16 + TB - 1) / TB, TB, 0, stream>>>(h, dinv, mbuf, n16);
        spmm_edges_kernel<<<(ethreads + TB - 1) / TB, TB, 0, stream>>>(row, col, dinv, h, mbuf, E);
        float beta = logf(0.5f / (float)(i + 1) + 1.0f);
        gemm_wmma_kernel<4, 2, true, true, false, true><<<gblocks, TB, 0, stream>>>(
            mbuf, x0, 0.9f, 0.1f, 64, N, convw + (size_t)i * 64 * 64, 64,
            nullptr, 1.0f - beta, beta, hbuf, 64, 64);
        h = hbuf;
    }

    // 5) out = h @ fc1_w + fc1_b   (40 cols padded to 48, masked store)
    gemm_wmma_kernel<3, 2, false, false, true, false><<<gblocks, TB, 0, stream>>>(
        h, nullptr, 1.0f, 0.0f, 64, N, fc1_w, 40, fc1_b,
        0.0f, 1.0f, out, 40, 40);

    (void)n_in; (void)out_size; (void)ws_size;
}
